// AddMM_30700426232147
// MI455X (gfx1250) — matrix operand order repaired and checked by reading, not compiled, not run on hardware
//
#include <hip/hip_runtime.h>

#define N_MODELS 16
#define IN_F     1024
#define OUT_F    1024
#define N_TOK    8192
#define MAX_T16  (N_TOK / 16 + N_MODELS)    // 528 padded 16-row tiles (fallback)
#define MAX_T64  (N_TOK / 64 + N_MODELS)    // 144 padded 64-row tiles (fast)

typedef __attribute__((ext_vector_type(16))) __bf16 v16bf;
typedef __attribute__((ext_vector_type(8)))  float  v8f;
typedef __attribute__((ext_vector_type(4)))  int    v4i32;

// Async global->LDS staging (CDNA5): use if the toolchain exposes it.
#if defined(__has_builtin)
#if __has_builtin(__builtin_amdgcn_global_load_async_to_lds_b128) && \
    __has_builtin(__builtin_amdgcn_s_wait_asynccnt)
#define USE_ASYNC_LDS 1
#endif
#endif
#ifndef USE_ASYNC_LDS
#define USE_ASYNC_LDS 0
#endif

#define AS1 __attribute__((address_space(1)))
#define AS3 __attribute__((address_space(3)))

// ---- workspace layout (ints) ----
#define WS_COUNTS 0      // 16
#define WS_OFFS   16     // 17  token offsets (exclusive scan, +total)
#define WS_CURS   33     // 16  scatter cursors
#define WS_TOFF16 49     // 17  16-row tile offsets
#define WS_TOFF64 66     // 17  64-row tile offsets
#define WS_PERM   83     // N_TOK permutation (tokens grouped by expert)

#define OFF_XHI  ((size_t)65536)                           // 8192x1024 bf16 = 16MB
#define OFF_XLO  (OFF_XHI + (size_t)16 * 1024 * 1024)      // 16MB
#define OFF_WHI  (OFF_XLO + (size_t)16 * 1024 * 1024)      // fragment-ready bf16 hi = 32MB
#define OFF_WLO  (OFF_WHI + (size_t)32 * 1024 * 1024)      // fragment-ready bf16 lo = 32MB
#define WS_REQ   (OFF_WLO + (size_t)32 * 1024 * 1024)

__device__ __forceinline__ unsigned short f2bf_rne(float f) {
  unsigned u = __float_as_uint(f);
  u += 0x7FFFu + ((u >> 16) & 1u);       // round to nearest even
  return (unsigned short)(u >> 16);
}
__device__ __forceinline__ float bf2f(unsigned short h) {
  return __uint_as_float(((unsigned)h) << 16);
}
__device__ __forceinline__ void split_bf(float f, unsigned short& hi, unsigned short& lo) {
  hi = f2bf_rne(f);
  lo = f2bf_rne(f - bf2f(hi));
}

// ======================= routing =======================
__global__ void moe_init(int* ws) {
  int t = threadIdx.x;
  if (t < WS_PERM) ws[t] = 0;
}

__global__ void moe_hist(const int* __restrict__ idxs, int* ws) {
  int t = blockIdx.x * blockDim.x + threadIdx.x;
  atomicAdd(&ws[WS_COUNTS + idxs[t]], 1);
}

__global__ void moe_scan(int* ws) {
  if (threadIdx.x == 0) {
    int acc = 0, t16 = 0, t64 = 0;
    for (int e = 0; e < N_MODELS; ++e) {
      ws[WS_OFFS + e]   = acc;
      ws[WS_TOFF16 + e] = t16;
      ws[WS_TOFF64 + e] = t64;
      int c = ws[WS_COUNTS + e];
      acc += c;
      t16 += (c + 15) >> 4;
      t64 += (c + 63) >> 6;
    }
    ws[WS_OFFS + N_MODELS]   = acc;
    ws[WS_TOFF16 + N_MODELS] = t16;
    ws[WS_TOFF64 + N_MODELS] = t64;
  }
}

__global__ void moe_scatter(const int* __restrict__ idxs, int* ws) {
  int t = blockIdx.x * blockDim.x + threadIdx.x;
  int e = idxs[t];
  int pos = atomicAdd(&ws[WS_CURS + e], 1);
  ws[WS_PERM + ws[WS_OFFS + e] + pos] = t;
}

// ======================= one-shot split-bf16 preprocessing =======================
__global__ void moe_conv_x(const float* __restrict__ x,
                           unsigned* __restrict__ xhi32,
                           unsigned* __restrict__ xlo32) {
  int t = blockIdx.x * blockDim.x + threadIdx.x;   // one dword pair per thread
  float2 f = ((const float2*)x)[t];
  unsigned short h0, l0, h1, l1;
  split_bf(f.x, h0, l0);
  split_bf(f.y, h1, l1);
  xhi32[t] = (unsigned)h0 | ((unsigned)h1 << 16);
  xlo32[t] = (unsigned)l0 | ((unsigned)l1 << 16);
}

// w -> WMMA-B-fragment-ready layout: for (expert, 32-K chunk, column n) the
// 16 K-pair dwords are contiguous, so a lane's 8-dword fragment is two
// contiguous global_load_b128 straight into WMMA source registers.
// dword index = ((e*32 + kchunk)*1024 + n)*16 + j
__global__ void moe_conv_w(const float* __restrict__ w,
                           uint4* __restrict__ whi4,
                           uint4* __restrict__ wlo4) {
  int t = blockIdx.x * blockDim.x + threadIdx.x;   // 16*32*1024 threads
  int n      = t & (OUT_F - 1);
  int kchunk = (t >> 10) & 31;
  int e      = t >> 15;
  const float* src = w + (size_t)e * IN_F * OUT_F + (size_t)(kchunk * 32) * OUT_F + n;
  size_t obase = (size_t)t * 4;
#pragma unroll
  for (int g = 0; g < 4; ++g) {
    unsigned hv[4], lv[4];
#pragma unroll
    for (int jj = 0; jj < 4; ++jj) {
      int j = g * 4 + jj;
      float f0 = src[(2 * j + 0) * OUT_F];
      float f1 = src[(2 * j + 1) * OUT_F];
      unsigned short h0, l0, h1, l1;
      split_bf(f0, h0, l0);
      split_bf(f1, h1, l1);
      hv[jj] = (unsigned)h0 | ((unsigned)h1 << 16);
      lv[jj] = (unsigned)l0 | ((unsigned)l1 << 16);
    }
    uint4 H; H.x = hv[0]; H.y = hv[1]; H.z = hv[2]; H.w = hv[3];
    uint4 L; L.x = lv[0]; L.y = lv[1]; L.z = lv[2]; L.w = lv[3];
    whi4[obase + g] = H;
    wlo4[obase + g] = L;
  }
}

// ======================= shared GEMM helpers =======================
union Frag { unsigned u[8]; v16bf v; };

__device__ __forceinline__ void frag_from(Frag& f, uint4 a, uint4 b) {
  f.u[0] = a.x; f.u[1] = a.y; f.u[2] = a.z; f.u[3] = a.w;
  f.u[4] = b.x; f.u[5] = b.y; f.u[6] = b.z; f.u[7] = b.w;
}

__device__ __forceinline__ bool tile_lookup(const int* __restrict__ ws,
                                            int* sMeta, int* sTok,
                                            int toffBase, int tileM) {
  const int tid = threadIdx.x;
  if (tid == 0) {
    int ty = blockIdx.y;
    if (ty >= ws[toffBase + N_MODELS]) {
      sMeta[0] = -1;
    } else {
      int e = 0;
      while (ty >= ws[toffBase + e + 1]) ++e;
      int lt = ty - ws[toffBase + e];
      int rowStart = ws[WS_OFFS + e] + lt * tileM;
      int valid = ws[WS_OFFS + e + 1] - rowStart;
      if (valid > tileM) valid = tileM;
      sMeta[0] = e; sMeta[1] = rowStart; sMeta[2] = valid;
    }
  }
  __syncthreads();
  if (sMeta[0] < 0) return false;                  // uniform for whole block
  if (tid < tileM) {
    int m = (tid < sMeta[2]) ? tid : 0;            // pad tile with row 0
    sTok[tid] = ws[WS_PERM + sMeta[1] + m];
  }
  __syncthreads();
  return true;
}

// Store one 16x16 C tile whose rows sit at [mOff, mOff+16) of the row tile.
__device__ __forceinline__ void store_tile(const float* __restrict__ bias,
                                           float* __restrict__ out,
                                           const int* sTok, int expert, int valid,
                                           int lane, int mOff, int col, v8f acc) {
  const float bv  = bias[expert * OUT_F + col];
  const int mBase = mOff + ((lane < 16) ? 0 : 8); // C layout: VGPR r -> M = mBase+r
  if (valid >= mOff + 16) {                        // fully-valid subtile
#pragma unroll
    for (int r = 0; r < 8; ++r) {
      float y = acc[r] + bv;
      out[(size_t)sTok[mBase + r] * OUT_F + col] = (y > 0.0f) ? y : 0.0f;
    }
  } else {
#pragma unroll
    for (int r = 0; r < 8; ++r) {
      int m = mBase + r;
      if (m < valid) {
        float y = acc[r] + bv;
        out[(size_t)sTok[m] * OUT_F + col] = (y > 0.0f) ? y : 0.0f;
      }
    }
  }
}

// ======================= fast GEMM =======================
// One block: 64 tokens x 128 output cols; 8 wave32s in a 2(M) x 4(N) grid;
// each wave computes a 32x32 C slab (4 accumulators, 12 WMMAs per K-chunk,
// A fragments shared across N, B fragments shared across M).
__global__ void __launch_bounds__(256) moe_gemm_fast(
    const unsigned* __restrict__ xhi32, const unsigned* __restrict__ xlo32,
    const uint4* __restrict__ whi4, const uint4* __restrict__ wlo4,
    const float* __restrict__ bias, const int* __restrict__ ws,
    float* __restrict__ out) {
  __shared__ int sMeta[3];
  __shared__ int sTok[64];
  __shared__ __align__(16) unsigned sAhi[64 * 16];   // 64 tokens x 32 bf16 (hi)
  __shared__ __align__(16) unsigned sAlo[64 * 16];   // 64 tokens x 32 bf16 (lo)

  if (!tile_lookup(ws, sMeta, sTok, WS_TOFF64, 64)) return;
  const int expert = sMeta[0];
  const int valid  = sMeta[2];

  const int tid    = threadIdx.x;
  const int lane   = tid & 31;
  const int waveId = tid >> 5;
  const int mGrp   = (waveId >> 2) * 32;           // wave's token-row group
  const int col0   = blockIdx.x * 128 + (waveId & 3) * 32 + (lane & 15);
  const int col1   = col0 + 16;
  const int half   = lane >> 4;                    // K-half owned by this lane

  // B fragment base (uint4 units): ((e*32+kchunk)*1024 + col)*4 + half*2
  size_t bIdx0 = (((size_t)expert * 32) * 1024 + col0) * 4 + half * 2;
  size_t bIdx1 = bIdx0 + 16 * 4;

  // A staging: thread t moves one b128 (4 dwords) of row t>>2, segment t&3
  const int sRow = tid >> 2;
  const int sSeg = (tid & 3) * 4;
  const size_t xb   = (size_t)sTok[sRow] * 512 + sSeg;
  const int    sIdx = sRow * 16 + sSeg;

  v8f acc00 = {}, acc01 = {}, acc10 = {}, acc11 = {};

  for (int kBase = 0; kBase < IN_F; kBase += 32) {
    const int kp0 = kBase >> 1;
    // ---- stage A tile ----
#if USE_ASYNC_LDS
    // CDNA5 async global->LDS: no VGPR round trip, tracked by ASYNCcnt
    __builtin_amdgcn_global_load_async_to_lds_b128(
        (AS1 v4i32*)&xhi32[xb + kp0], (AS3 v4i32*)&sAhi[sIdx], 0, 0);
    __builtin_amdgcn_global_load_async_to_lds_b128(
        (AS1 v4i32*)&xlo32[xb + kp0], (AS3 v4i32*)&sAlo[sIdx], 0, 0);
    __builtin_amdgcn_s_wait_asynccnt(0);
#else
    *(uint4*)&sAhi[sIdx] = *(const uint4*)&xhi32[xb + kp0];
    *(uint4*)&sAlo[sIdx] = *(const uint4*)&xlo32[xb + kp0];
#endif
    __syncthreads();

    // ---- A fragments for both 16-row subtiles (contiguous k = 16*half + s) ----
    Frag aHi0, aLo0, aHi1, aLo1;
    {
      const int row = mGrp + (lane & 15);
      const uint4* pH0 = (const uint4*)(&sAhi[row * 16]);
      const uint4* pL0 = (const uint4*)(&sAlo[row * 16]);
      const uint4* pH1 = (const uint4*)(&sAhi[(row + 16) * 16]);
      const uint4* pL1 = (const uint4*)(&sAlo[(row + 16) * 16]);
      frag_from(aHi0, pH0[2 * half], pH0[2 * half + 1]);
      frag_from(aLo0, pL0[2 * half], pL0[2 * half + 1]);
      frag_from(aHi1, pH1[2 * half], pH1[2 * half + 1]);
      frag_from(aLo1, pL1[2 * half], pL1[2 * half + 1]);
    }

    // ---- B fragments: contiguous b128 pairs, shared by both M subtiles ----
    Frag bHi0, bLo0, bHi1, bLo1;
    frag_from(bHi0, whi4[bIdx0], whi4[bIdx0 + 1]);
    frag_from(bLo0, wlo4[bIdx0], wlo4[bIdx0 + 1]);
    frag_from(bHi1, whi4[bIdx1], whi4[bIdx1 + 1]);
    frag_from(bLo1, wlo4[bIdx1], wlo4[bIdx1 + 1]);
    bIdx0 += (size_t)1024 * 4;
    bIdx1 += (size_t)1024 * 4;

    // ---- bf16x3 emulated-f32 MACs: 2 M-subtiles x 2 N-tiles ----
    acc00 = __builtin_amdgcn_wmma_f32_16x16x32_bf16(false, aHi0.v, false, bHi0.v, (short)0, acc00, false, false);
    acc01 = __builtin_amdgcn_wmma_f32_16x16x32_bf16(false, aHi0.v, false, bHi1.v, (short)0, acc01, false, false);
    acc10 = __builtin_amdgcn_wmma_f32_16x16x32_bf16(false, aHi1.v, false, bHi0.v, (short)0, acc10, false, false);
    acc11 = __builtin_amdgcn_wmma_f32_16x16x32_bf16(false, aHi1.v, false, bHi1.v, (short)0, acc11, false, false);
    acc00 = __builtin_amdgcn_wmma_f32_16x16x32_bf16(false, aHi0.v, false, bLo0.v, (short)0, acc00, false, false);
    acc01 = __builtin_amdgcn_wmma_f32_16x16x32_bf16(false, aHi0.v, false, bLo1.v, (short)0, acc01, false, false);
    acc10 = __builtin_amdgcn_wmma_f32_16x16x32_bf16(false, aHi1.v, false, bLo0.v, (short)0, acc10, false, false);
    acc11 = __builtin_amdgcn_wmma_f32_16x16x32_bf16(false, aHi1.v, false, bLo1.v, (short)0, acc11, false, false);
    acc00 = __builtin_amdgcn_wmma_f32_16x16x32_bf16(false, aLo0.v, false, bHi0.v, (short)0, acc00, false, false);
    acc01 = __builtin_amdgcn_wmma_f32_16x16x32_bf16(false, aLo0.v, false, bHi1.v, (short)0, acc01, false, false);
    acc10 = __builtin_amdgcn_wmma_f32_16x16x32_bf16(false, aLo1.v, false, bHi0.v, (short)0, acc10, false, false);
    acc11 = __builtin_amdgcn_wmma_f32_16x16x32_bf16(false, aLo1.v, false, bHi1.v, (short)0, acc11, false, false);
    __syncthreads();
  }

  store_tile(bias, out, sTok, expert, valid, lane, mGrp,      col0, acc00);
  store_tile(bias, out, sTok, expert, valid, lane, mGrp,      col1, acc01);
  store_tile(bias, out, sTok, expert, valid, lane, mGrp + 16, col0, acc10);
  store_tile(bias, out, sTok, expert, valid, lane, mGrp + 16, col1, acc11);
}

// ======================= fallback GEMM: in-loop conversion =======================
__global__ void __launch_bounds__(256) moe_gemm_fb(
    const float* __restrict__ x, const float* __restrict__ w,
    const float* __restrict__ bias, const int* __restrict__ ws,
    float* __restrict__ out) {
  __shared__ int sMeta[3];
  __shared__ int sTok[16];
  __shared__ __align__(16) unsigned short sAhi[16 * 32];
  __shared__ __align__(16) unsigned short sAlo[16 * 32];

  if (!tile_lookup(ws, sMeta, sTok, WS_TOFF16, 16)) return;
  const int expert = sMeta[0];
  const int valid  = sMeta[2];

  const int tid    = threadIdx.x;
  const int lane   = tid & 31;
  const int waveId = tid >> 5;
  const int col    = blockIdx.x * 128 + waveId * 16 + (lane & 15);
  const int kOffB  = (lane < 16) ? 0 : 16;
  const float* wExp = w + (size_t)expert * IN_F * OUT_F;

  const int aM = (tid * 2) >> 5;
  const int aK = (tid * 2) & 31;
  const float* xRow = x + (size_t)sTok[aM] * IN_F + aK;

  v8f acc = {};

  for (int kBase = 0; kBase < IN_F; kBase += 32) {
    {
      float f0 = xRow[kBase + 0];
      float f1 = xRow[kBase + 1];
      unsigned short h0, l0, h1, l1;
      split_bf(f0, h0, l0);
      split_bf(f1, h1, l1);
      sAhi[aM * 32 + aK + 0] = h0;
      sAhi[aM * 32 + aK + 1] = h1;
      sAlo[aM * 32 + aK + 0] = l0;
      sAlo[aM * 32 + aK + 1] = l1;
    }
    __syncthreads();

    Frag aHi, aLo;
    {
      const int row  = lane & 15;
      const int half = lane >> 4;
      const uint4* pH = (const uint4*)(&sAhi[row * 32]);
      const uint4* pL = (const uint4*)(&sAlo[row * 32]);
      frag_from(aHi, pH[2 * half], pH[2 * half + 1]);
      frag_from(aLo, pL[2 * half], pL[2 * half + 1]);
    }

    Frag bHi, bLo;
    {
      const float* wp = wExp + (size_t)(kBase + kOffB) * OUT_F + col;
#pragma unroll
      for (int v = 0; v < 8; ++v) {
        float f0 = wp[(2 * v + 0) * OUT_F];
        float f1 = wp[(2 * v + 1) * OUT_F];
        unsigned short h0, l0, h1, l1;
        split_bf(f0, h0, l0);
        split_bf(f1, h1, l1);
        bHi.u[v] = (unsigned)h0 | ((unsigned)h1 << 16);
        bLo.u[v] = (unsigned)l0 | ((unsigned)l1 << 16);
      }
    }

    acc = __builtin_amdgcn_wmma_f32_16x16x32_bf16(false, aHi.v, false, bHi.v,
                                                  (short)0, acc, false, false);
    acc = __builtin_amdgcn_wmma_f32_16x16x32_bf16(false, aHi.v, false, bLo.v,
                                                  (short)0, acc, false, false);
    acc = __builtin_amdgcn_wmma_f32_16x16x32_bf16(false, aLo.v, false, bHi.v,
                                                  (short)0, acc, false, false);
    __syncthreads();
  }

  store_tile(bias, out, sTok, expert, valid, lane, 0, col, acc);
}

extern "C" void kernel_launch(void* const* d_in, const int* in_sizes, int n_in,
                              void* d_out, int out_size, void* d_ws, size_t ws_size,
                              hipStream_t stream) {
  const float* x    = (const float*)d_in[0];
  const int*   idxs = (const int*)d_in[1];
  const float* w    = (const float*)d_in[2];
  const float* b    = (const float*)d_in[3];
  float* out = (float*)d_out;
  int*   ws  = (int*)d_ws;

  moe_init<<<1, 128, 0, stream>>>(ws);
  moe_hist<<<N_TOK / 256, 256, 0, stream>>>(idxs, ws);
  moe_scan<<<1, 32, 0, stream>>>(ws);
  moe_scatter<<<N_TOK / 256, 256, 0, stream>>>(idxs, ws);

  if (ws_size >= WS_REQ) {
    unsigned* xhi32 = (unsigned*)((char*)d_ws + OFF_XHI);
    unsigned* xlo32 = (unsigned*)((char*)d_ws + OFF_XLO);
    uint4*    whi4  = (uint4*)((char*)d_ws + OFF_WHI);
    uint4*    wlo4  = (uint4*)((char*)d_ws + OFF_WLO);
    moe_conv_x<<<(N_TOK * IN_F / 2) / 256, 256, 0, stream>>>(x, xhi32, xlo32);
    moe_conv_w<<<(N_MODELS * 32 * OUT_F) / 256, 256, 0, stream>>>(w, whi4, wlo4);
    dim3 grid(OUT_F / 128, MAX_T64);
    moe_gemm_fast<<<grid, 256, 0, stream>>>(xhi32, xlo32, whi4, wlo4, b, ws, out);
  } else {
    dim3 grid(OUT_F / 128, MAX_T16);
    moe_gemm_fb<<<grid, 256, 0, stream>>>(x, w, b, ws, out);
  }
}
